// MultiHeadSelfAttentionLayer_21930103013454
// MI455X (gfx1250) — matrix operand order repaired and checked by reading, not compiled, not run on hardware
//
#include <hip/hip_runtime.h>
#include <hip/hip_bf16.h>

// ---------------------------------------------------------------------------
// MultiHeadSelfAttentionLayer for MI455X (gfx1250, wave32, WMMA bf16)
//
// Shapes: N=4, L=2048, E=H=1024, heads=16, dh=64. The reference's raw
// reshape makes each (n,head) a contiguous 2048x64 tile of Q/K/V.
// Softmax is over the query axis (per-column), and V is only scaled by
// the row-sums of A, so we never materialize the 4x16x2048x2048 tensor:
//   pass1: colmax[b], colsum[b] (online, per column)
//   pass2: R[a] = sum_b exp(S[a,b]-colmax[b]) / colsum[b]
// All GEMMs run on v_wmma_f32_16x16x32_bf16 (bf16 in, fp32 accumulate).
// Projection GEMMs stage tiles in LDS via the CDNA5 async global->LDS
// engine (ASYNCcnt) when the toolchain exposes the builtin.
// ---------------------------------------------------------------------------

typedef __bf16 bf16;
typedef __attribute__((ext_vector_type(16))) bf16  v16bf;
typedef __attribute__((ext_vector_type(8)))  bf16  v8bf;
typedef __attribute__((ext_vector_type(8)))  float v8f;
typedef int v4i32 __attribute__((vector_size(16)));   // matches builtin's b128 pointee

#define NB    4
#define LSEQ  2048
#define EDIM  1024
#define HDIM  1024
#define NHEAD 16
#define DH    64
#define NL    (NB * LSEQ)    /* 8192 rows for the projection GEMMs */
#define NE    (NB * NHEAD)   /* 64 (n,head) pairs */

#if defined(__has_builtin)
#if __has_builtin(__builtin_amdgcn_global_load_async_to_lds_b128)
#define HAVE_ASYNC_LDS 1
#endif
#endif
#ifndef HAVE_ASYNC_LDS
#define HAVE_ASYNC_LDS 0
#endif

// LDS tile row stride (elements): 32 data + 16 pad -> 96B rows (32B aligned,
// spreads fragment rows across LDS banks).
#define LDS_STRIDE 48

// ---- fragment loaders, contiguous k order (k = 16h + s) for both A and B ----------

__device__ __forceinline__ v16bf cat8(v8bf lo, v8bf hi) {
  return __builtin_shufflevector(lo, hi, 0,1,2,3,4,5,6,7,8,9,10,11,12,13,14,15);
}

// A matrix 16x32 (MxK), row-major source with leading dim ld.
// lane<16: K kk..kk+15; lane>=16: K kk+16..kk+31; M = lane&15.
__device__ __forceinline__ v16bf load_a_frag(const bf16* base,
                                             int row, int ld, int kk, int lane) {
  const bf16* p = base + (size_t)row * ld + kk + ((lane & 16) ? 16 : 0);
  return *(const v16bf*)p;
}

// B matrix 32x16 (KxN). Source is B^T row-major (row = N column, ld = K dim),
// so each lane reads 16 contiguous bf16: lane<16 -> K kk..kk+15, lane>=16 -> kk+16..kk+31.
__device__ __forceinline__ v16bf load_b_frag(const bf16* baseT,
                                             int col, int ld, int kk, int lane) {
  const bf16* p = baseT + (size_t)col * ld + kk + ((lane & 16) ? 16 : 0);
  return *(const v16bf*)p;
}

__device__ __forceinline__ v8f wmma_bf16(v16bf a, v16bf b, v8f c) {
  return __builtin_amdgcn_wmma_f32_16x16x32_bf16(false, a, false, b, (short)0, c,
                                                 false, false);
}

#if HAVE_ASYNC_LDS
// Async-copy a 128-row x 32-col bf16 slice (global, leading dim gld) into an
// LDS tile with row stride LDS_STRIDE. 512 16B chunks, 2 per thread.
__device__ __forceinline__ void async_copy_tile(const bf16* gsrc, int grow0,
                                                int gld, int kk, bf16* lds,
                                                int tid) {
#pragma unroll
  for (int it = 0; it < 2; ++it) {
    int chunk = tid + it * 256;           // 0..511
    int r = chunk >> 2;                   // row 0..127
    int c = chunk & 3;                    // 16B chunk within 64B row
    const bf16* g = gsrc + (size_t)(grow0 + r) * gld + kk + c * 8;
    bf16* l = lds + r * LDS_STRIDE + c * 8;
    __builtin_amdgcn_global_load_async_to_lds_b128(
        (__attribute__((address_space(1))) v4i32*)(void*)g,
        (__attribute__((address_space(3))) v4i32*)(void*)l, 0, 0);
  }
}
#endif

// ---- precision conversion --------------------------------------------------

__global__ void __launch_bounds__(256)
cvt_f32_bf16(const float* __restrict__ src, bf16* __restrict__ dst, int n) {
  int i = blockIdx.x * 256 + threadIdx.x;
  if (i < n) dst[i] = (bf16)src[i];
}

// Transpose-convert the four 1024x1024 weight matrices: Wt[h][e] = W[e][h].
__global__ void __launch_bounds__(256)
cvt_w_transpose(const float* __restrict__ Wq, const float* __restrict__ Wk,
                const float* __restrict__ Wv, const float* __restrict__ Wo,
                bf16* __restrict__ Wtq, bf16* __restrict__ Wtk,
                bf16* __restrict__ Wtv, bf16* __restrict__ Wto) {
  const float* src = (blockIdx.z == 0) ? Wq : (blockIdx.z == 1) ? Wk
                   : (blockIdx.z == 2) ? Wv : Wo;
  bf16* dst = (blockIdx.z == 0) ? Wtq : (blockIdx.z == 1) ? Wtk
            : (blockIdx.z == 2) ? Wtv : Wto;
  int h = blockIdx.x * 16 + (threadIdx.x & 15);
  int e = blockIdx.y * 16 + (threadIdx.x >> 4);
  dst[(size_t)h * EDIM + e] = (bf16)src[(size_t)e * HDIM + h];
}

// ---- tiled WMMA GEMM: D = (A[MxK] * B[KxN]) + bias, optional scale ---------
// A: bf16 row-major [M x 1024]; Bt: bf16 B^T row-major [1024 x 1024].
// Block = 256 thr = 8 waves, tile 128x128, wave tile 32x64, K-step 32.
// Async path: double-buffered LDS staging of both A and B k-slices.

template <bool OUTF32>
__global__ void __launch_bounds__(256)
gemm_bias(const bf16* __restrict__ A, const bf16* __restrict__ Bt,
          const float* __restrict__ bias, void* __restrict__ dstv, float scale) {
  const int K = 1024, NCOL = 1024;
  int tid  = threadIdx.x;
  int lane = tid & 31;
  int w    = tid >> 5;
  int wy   = w >> 1, wx = w & 1;
  int bm0 = blockIdx.x * 128;
  int bn0 = blockIdx.y * 128;
  int m0 = bm0 + wy * 32;
  int n0 = bn0 + wx * 64;

  v8f acc[2][4];
#pragma unroll
  for (int i = 0; i < 2; ++i)
#pragma unroll
    for (int j = 0; j < 4; ++j)
#pragma unroll
      for (int r = 0; r < 8; ++r) acc[i][j][r] = 0.0f;

#if HAVE_ASYNC_LDS
  __shared__ bf16 lsA[2][128 * LDS_STRIDE];
  __shared__ bf16 lsB[2][128 * LDS_STRIDE];

  async_copy_tile(A,  bm0, K, 0, lsA[0], tid);
  async_copy_tile(Bt, bn0, K, 0, lsB[0], tid);
  int buf = 0;
  for (int kk = 0; kk < K; kk += 32) {
    if (kk + 32 < K) {
      async_copy_tile(A,  bm0, K, kk + 32, lsA[buf ^ 1], tid);
      async_copy_tile(Bt, bn0, K, kk + 32, lsB[buf ^ 1], tid);
      asm volatile("s_wait_asynccnt 0x4" ::: "memory");  // current slice done
    } else {
      asm volatile("s_wait_asynccnt 0x0" ::: "memory");
    }
    __syncthreads();

    v16bf af[2], bfg[4];
#pragma unroll
    for (int i = 0; i < 2; ++i)
      af[i] = load_a_frag(lsA[buf], wy * 32 + i * 16 + (lane & 15),
                          LDS_STRIDE, 0, lane);
#pragma unroll
    for (int j = 0; j < 4; ++j)
      bfg[j] = load_b_frag(lsB[buf], wx * 64 + j * 16 + (lane & 15),
                           LDS_STRIDE, 0, lane);
#pragma unroll
    for (int i = 0; i < 2; ++i)
#pragma unroll
      for (int j = 0; j < 4; ++j)
        acc[i][j] = wmma_bf16(af[i], bfg[j], acc[i][j]);

    __syncthreads();
    buf ^= 1;
  }
#else
  for (int kk = 0; kk < K; kk += 32) {
    v16bf af[2], bfg[4];
#pragma unroll
    for (int i = 0; i < 2; ++i)
      af[i] = load_a_frag(A, m0 + i * 16 + (lane & 15), K, kk, lane);
#pragma unroll
    for (int j = 0; j < 4; ++j)
      bfg[j] = load_b_frag(Bt, n0 + j * 16 + (lane & 15), K, kk, lane);
#pragma unroll
    for (int i = 0; i < 2; ++i)
#pragma unroll
      for (int j = 0; j < 4; ++j)
        acc[i][j] = wmma_bf16(af[i], bfg[j], acc[i][j]);
  }
#endif

#pragma unroll
  for (int j = 0; j < 4; ++j) {
    int col = n0 + j * 16 + (lane & 15);
    float b = bias[col];
#pragma unroll
    for (int i = 0; i < 2; ++i) {
      int rbase = m0 + i * 16 + ((lane & 16) ? 8 : 0);
#pragma unroll
      for (int r = 0; r < 8; ++r) {
        float v = (acc[i][j][r] + b) * scale;
        if (OUTF32)
          ((float*)dstv)[(size_t)(rbase + r) * NCOL + col] = v;
        else
          ((bf16*)dstv)[(size_t)(rbase + r) * NCOL + col] = (bf16)v;
      }
    }
  }
}

// ---- pass 1: per-column (softmax over query axis) running max / sum --------
// Grid: (L/128, NE). Each wave owns 16 columns, loops over all 2048 rows.
// D-tile layout: column n lives in lanes n and n+16 -> final shfl_xor(16) merge.

__global__ void __launch_bounds__(256)
attn_colstats(const bf16* __restrict__ Qb, const bf16* __restrict__ Kb,
              float* __restrict__ colmax, float* __restrict__ colsum) {
  int lane = threadIdx.x & 31;
  int w    = threadIdx.x >> 5;
  int ne   = blockIdx.y;
  int cb   = blockIdx.x * 128 + w * 16;
  const bf16* Qh = Qb + (size_t)ne * LSEQ * DH;
  const bf16* Kh = Kb + (size_t)ne * LSEQ * DH;

  v16bf bfg[2];
#pragma unroll
  for (int t = 0; t < 2; ++t)
    bfg[t] = load_b_frag(Kh, cb + (lane & 15), DH, t * 32, lane);

  float m = -__builtin_inff();
  float s = 0.0f;

  for (int a0 = 0; a0 < LSEQ; a0 += 32) {
    v8f acc[2];
#pragma unroll
    for (int i = 0; i < 2; ++i)
#pragma unroll
      for (int r = 0; r < 8; ++r) acc[i][r] = 0.0f;
#pragma unroll
    for (int i = 0; i < 2; ++i)
#pragma unroll
      for (int t = 0; t < 2; ++t) {
        v16bf af = load_a_frag(Qh, a0 + i * 16 + (lane & 15), DH, t * 32, lane);
        acc[i] = wmma_bf16(af, bfg[t], acc[i]);
      }
#pragma unroll
    for (int i = 0; i < 2; ++i) {
      float tmax = acc[i][0];
#pragma unroll
      for (int r = 1; r < 8; ++r) tmax = fmaxf(tmax, acc[i][r]);
      float m2 = fmaxf(m, tmax);
      float part = 0.0f;
#pragma unroll
      for (int r = 0; r < 8; ++r) part += __expf(acc[i][r] - m2);
      s = s * __expf(m - m2) + part;
      m = m2;
    }
  }

  float om = __shfl_xor(m, 16, 32);
  float os = __shfl_xor(s, 16, 32);
  float M = fmaxf(m, om);
  float S = s * __expf(m - M) + os * __expf(om - M);
  if (lane < 16) {
    colmax[ne * LSEQ + cb + lane] = M;
    colsum[ne * LSEQ + cb + lane] = S;
  }
}

// ---- pass 2: R[a] = sum_b exp(S[a,b]-colmax[b]) / colsum[b] ----------------
// Grid: (L/256, NE). Each wave owns 32 rows (A-frags register-resident),
// loops over all 2048 columns.

__global__ void __launch_bounds__(256)
attn_rowsum(const bf16* __restrict__ Qb, const bf16* __restrict__ Kb,
            const float* __restrict__ colmax, const float* __restrict__ colsum,
            float* __restrict__ Rrow) {
  int lane = threadIdx.x & 31;
  int w    = threadIdx.x >> 5;
  int ne   = blockIdx.y;
  int r0   = blockIdx.x * 256 + w * 32;
  const bf16* Qh = Qb + (size_t)ne * LSEQ * DH;
  const bf16* Kh = Kb + (size_t)ne * LSEQ * DH;

  v16bf af[2][2];
#pragma unroll
  for (int i = 0; i < 2; ++i)
#pragma unroll
    for (int t = 0; t < 2; ++t)
      af[i][t] = load_a_frag(Qh, r0 + i * 16 + (lane & 15), DH, t * 32, lane);

  v8f racc[2];
#pragma unroll
  for (int i = 0; i < 2; ++i)
#pragma unroll
    for (int r = 0; r < 8; ++r) racc[i][r] = 0.0f;

  for (int b0 = 0; b0 < LSEQ; b0 += 16) {
    v16bf bf0 = load_b_frag(Kh, b0 + (lane & 15), DH, 0, lane);
    v16bf bf1 = load_b_frag(Kh, b0 + (lane & 15), DH, 32, lane);
    v8f acc[2];
#pragma unroll
    for (int i = 0; i < 2; ++i)
#pragma unroll
      for (int r = 0; r < 8; ++r) acc[i][r] = 0.0f;
    acc[0] = wmma_bf16(af[0][0], bf0, acc[0]);
    acc[0] = wmma_bf16(af[0][1], bf1, acc[0]);
    acc[1] = wmma_bf16(af[1][0], bf0, acc[1]);
    acc[1] = wmma_bf16(af[1][1], bf1, acc[1]);

    int c = b0 + (lane & 15);
    float cm  = colmax[ne * LSEQ + c];
    float inv = 1.0f / colsum[ne * LSEQ + c];
#pragma unroll
    for (int i = 0; i < 2; ++i)
#pragma unroll
      for (int r = 0; r < 8; ++r)
        racc[i][r] += __expf(acc[i][r] - cm) * inv;
  }

  // reduce across each 16-lane half (rows stay separated by lane&16)
#pragma unroll
  for (int off = 1; off < 16; off <<= 1)
#pragma unroll
    for (int i = 0; i < 2; ++i)
#pragma unroll
      for (int r = 0; r < 8; ++r)
        racc[i][r] += __shfl_xor(racc[i][r], off, 32);

  if ((lane & 15) == 0) {
    int rowoff = (lane & 16) ? 8 : 0;
#pragma unroll
    for (int i = 0; i < 2; ++i)
#pragma unroll
      for (int r = 0; r < 8; ++r)
        Rrow[ne * LSEQ + r0 + i * 16 + rowoff + r] = racc[i][r];
  }
}

// ---- elementwise: AttnOut = R[row] * V  (bf16 out for final GEMM) ----------

__global__ void __launch_bounds__(256)
apply_rowscale(const bf16* __restrict__ Vb, const float* __restrict__ Rrow,
               bf16* __restrict__ Ab, int n) {
  int i = blockIdx.x * 256 + threadIdx.x;
  if (i < n) Ab[i] = (bf16)((float)Vb[i] * Rrow[i >> 6]);
}

// ---------------------------------------------------------------------------

extern "C" void kernel_launch(void* const* d_in, const int* in_sizes, int n_in,
                              void* d_out, int out_size, void* d_ws, size_t ws_size,
                              hipStream_t stream) {
  const float* X  = (const float*)d_in[0];
  const float* Wq = (const float*)d_in[1];
  const float* bq = (const float*)d_in[2];
  const float* Wk = (const float*)d_in[3];
  const float* bk = (const float*)d_in[4];
  const float* Wv = (const float*)d_in[5];
  const float* bv = (const float*)d_in[6];
  const float* Wo = (const float*)d_in[7];
  const float* bo = (const float*)d_in[8];
  float* out = (float*)d_out;

  char* ws = (char*)d_ws;
  size_t off = 0;
  bf16* Xb  = (bf16*)(ws + off); off += (size_t)NL * EDIM * sizeof(bf16);   // 16 MB
  bf16* Wtq = (bf16*)(ws + off); off += (size_t)EDIM * HDIM * sizeof(bf16); //  2 MB
  bf16* Wtk = (bf16*)(ws + off); off += (size_t)EDIM * HDIM * sizeof(bf16);
  bf16* Wtv = (bf16*)(ws + off); off += (size_t)EDIM * HDIM * sizeof(bf16);
  bf16* Wto = (bf16*)(ws + off); off += (size_t)EDIM * HDIM * sizeof(bf16);
  bf16* Qb  = (bf16*)(ws + off); off += (size_t)NL * HDIM * sizeof(bf16);   // 16 MB
  bf16* Kb  = (bf16*)(ws + off); off += (size_t)NL * HDIM * sizeof(bf16);
  bf16* Vb  = (bf16*)(ws + off); off += (size_t)NL * HDIM * sizeof(bf16);
  bf16* Ab  = (bf16*)(ws + off); off += (size_t)NL * HDIM * sizeof(bf16);
  float* colmax = (float*)(ws + off); off += (size_t)NE * LSEQ * sizeof(float);
  float* colsum = (float*)(ws + off); off += (size_t)NE * LSEQ * sizeof(float);
  float* Rrow   = (float*)(ws + off); off += (size_t)NE * LSEQ * sizeof(float);

  const int nX = NL * EDIM;                   // 8388608
  const float qk_scale = 1.0f / 32.0f;        // 1/sqrt(H) applied to Q and K each

  cvt_f32_bf16<<<nX / 256, 256, 0, stream>>>(X, Xb, nX);
  cvt_w_transpose<<<dim3(HDIM / 16, EDIM / 16, 4), 256, 0, stream>>>(
      Wq, Wk, Wv, Wo, Wtq, Wtk, Wtv, Wto);

  dim3 ggrid(NL / 128, HDIM / 128);           // (64, 8)
  gemm_bias<false><<<ggrid, 256, 0, stream>>>(Xb, Wtq, bq, Qb, qk_scale);
  gemm_bias<false><<<ggrid, 256, 0, stream>>>(Xb, Wtk, bk, Kb, qk_scale);
  gemm_bias<false><<<ggrid, 256, 0, stream>>>(Xb, Wtv, bv, Vb, 1.0f);

  attn_colstats<<<dim3(LSEQ / 128, NE), 256, 0, stream>>>(Qb, Kb, colmax, colsum);
  attn_rowsum<<<dim3(LSEQ / 256, NE), 256, 0, stream>>>(Qb, Kb, colmax, colsum, Rrow);

  apply_rowscale<<<nX / 256, 256, 0, stream>>>(Vb, Rrow, Ab, nX);

  gemm_bias<true><<<ggrid, 256, 0, stream>>>(Ab, Wto, bo, out, 1.0f);
}
